// Trans_GAT_60198261621556
// MI455X (gfx1250) — compile-verified, matrix operand order checked by reading, not run on hardware
//
#include <hip/hip_runtime.h>
#include <hip/hip_bf16.h>

#define NN     50000
#define EE     800000
#define FIN    128
#define HID    16
#define HEADS  8
#define H1DIM  128   // HEADS*HID
#define CC     16
#define NEG_SLOPE 0.2f

typedef __attribute__((ext_vector_type(16))) __bf16 v16bf;
typedef __attribute__((ext_vector_type(8)))  float  v8f;

// ---- monotone float<->int key for atomicMax-based segment max ----
__device__ __forceinline__ int   fkey(float f){ int i = __float_as_int(f); return i >= 0 ? i : (i ^ 0x7FFFFFFF); }
__device__ __forceinline__ float funkey(int k){ int i = (k >= 0) ? k : (k ^ 0x7FFFFFFF); return __int_as_float(i); }

// ---------------- fills ----------------
__global__ void fill_f32(float* p, float v, int n){
    int i = blockIdx.x * blockDim.x + threadIdx.x;
    if (i < n) p[i] = v;
}
__global__ void fill_i32(int* p, int v, int n){
    int i = blockIdx.x * blockDim.x + threadIdx.x;
    if (i < n) p[i] = v;
}

// ---------------- casts ----------------
__global__ void cast_bf16(const float* __restrict__ src, __bf16* __restrict__ dst, int n){
    int i = blockIdx.x * blockDim.x + threadIdx.x;
    if (i < n) dst[i] = (__bf16)src[i];
}
// W [K][Ncols] row-major  ->  Wt [Ncols][K] bf16 (so k-pairs are contiguous)
__global__ void transpose_cast_bf16(const float* __restrict__ W, __bf16* __restrict__ Wt, int K, int Ncols){
    int i = blockIdx.x * blockDim.x + threadIdx.x;
    if (i < K * Ncols){
        int k = i / Ncols, c = i % Ncols;
        Wt[(size_t)c * K + k] = (__bf16)W[i];
    }
}

// ---------------- WMMA GEMM: out[N,NC] = Abf[N,128] @ Wt[NC,128]^T ----------------
// grid.x = row tiles handled per block via 8 waves; each wave does one 16x16 tile.
__device__ __forceinline__ void wmma_tile_128(const __bf16* __restrict__ arow,
                                              const __bf16* __restrict__ brow,
                                              int hl, v8f& acc){
    #pragma unroll
    for (int kk = 0; kk < 128; kk += 32){
        v16bf a, b;
        #pragma unroll
        for (int i = 0; i < 8; ++i){
            int k = kk + ((i >> 2) << 4) + (hl << 3) + ((i & 3) << 1);
            a[2*i]   = arow[k];
            a[2*i+1] = arow[k+1];
            b[2*i]   = brow[k];
            b[2*i+1] = brow[k+1];
        }
        acc = __builtin_amdgcn_wmma_f32_16x16x32_bf16(false, a, false, b,
                                                      (short)0, acc, false, false);
    }
}

// C = A[N,128] @ Wt(128 cols)[128,128]^T : grid = N/16 blocks, 256 threads (8 waves = 8 col tiles)
__global__ void gemm_n128(const __bf16* __restrict__ A, const __bf16* __restrict__ Wt,
                          float* __restrict__ out){
    int wave = threadIdx.x >> 5;
    int lane = threadIdx.x & 31;
    int rl = lane & 15, hl = lane >> 4;
    int row0 = blockIdx.x * 16;
    int col0 = wave * 16;
    const __bf16* arow = A  + (size_t)(row0 + rl) * 128;
    const __bf16* brow = Wt + (size_t)(col0 + rl) * 128;
    v8f acc = {};
    wmma_tile_128(arow, brow, hl, acc);
    float* orow = out + (size_t)(row0 + hl * 8) * 128 + col0 + rl;
    #pragma unroll
    for (int r = 0; r < 8; ++r) orow[(size_t)r * 128] = acc[r];
}

// C = A[N,128] @ Wt(16 cols)[16,128]^T : 8 row tiles per block
__global__ void gemm_n16(const __bf16* __restrict__ A, const __bf16* __restrict__ Wt,
                         float* __restrict__ out, int ntiles){
    int wave = threadIdx.x >> 5;
    int lane = threadIdx.x & 31;
    int tile = blockIdx.x * 8 + wave;
    if (tile >= ntiles) return;            // whole wave exits together: EXEC all-1s for WMMA
    int rl = lane & 15, hl = lane >> 4;
    int row0 = tile * 16;
    const __bf16* arow = A  + (size_t)(row0 + rl) * 128;
    const __bf16* brow = Wt + (size_t)rl * 128;
    v8f acc = {};
    wmma_tile_128(arow, brow, hl, acc);
    float* orow = out + (size_t)(row0 + hl * 8) * 16 + rl;
    #pragma unroll
    for (int r = 0; r < 8; ++r) orow[(size_t)r * 16] = acc[r];
}

// ---------------- per-node attention coefficients ----------------
__global__ void alphas_k(const float* __restrict__ h, const float* __restrict__ a_src,
                         const float* __restrict__ a_dst, float* __restrict__ as,
                         float* __restrict__ ad, int H, int O){
    int t = blockIdx.x * blockDim.x + threadIdx.x;
    if (t >= NN * H) return;
    int node = t / H, hh = t % H;
    const float* hp = h + (size_t)node * H * O + hh * O;
    float s = 0.f, d = 0.f;
    for (int o = 0; o < O; ++o){
        s += hp[o] * a_src[hh * O + o];
        d += hp[o] * a_dst[hh * O + o];
    }
    as[t] = s; ad[t] = d;
}

// ---------------- edge pass 1: segment max (int-key atomicMax) ----------------
__global__ void edge_max_k(const int* __restrict__ ei, const float* __restrict__ as,
                           const float* __restrict__ ad, int* __restrict__ mk, int H){
    int t = blockIdx.x * blockDim.x + threadIdx.x;
    int total = (EE + NN) * H;
    if (t >= total) return;
    int e = t / H, hh = t % H;
    int s, d;
    if (e < EE){ s = ei[e]; d = ei[EE + e]; } else { s = d = e - EE; }
    float v = as[s * H + hh] + ad[d * H + hh];
    v = v > 0.f ? v : v * NEG_SLOPE;
    atomicMax(&mk[d * H + hh], fkey(v));
}

// ---------------- edge pass 2: exp weights + denom + unnormalized aggregation ----------------
__global__ void edge_acc_k(const int* __restrict__ ei, const float* __restrict__ as,
                           const float* __restrict__ ad, const int* __restrict__ mk,
                           const float* __restrict__ hsrc, float* __restrict__ den,
                           float* __restrict__ agg, int H, int O){
    int t = blockIdx.x * blockDim.x + threadIdx.x;
    int total = (EE + NN) * H;
    if (t >= total) return;
    int e = t / H, hh = t % H;
    int s, d;
    if (e < EE){ s = ei[e]; d = ei[EE + e]; } else { s = d = e - EE; }
    float v = as[s * H + hh] + ad[d * H + hh];
    v = v > 0.f ? v : v * NEG_SLOPE;
    float m = funkey(mk[d * H + hh]);
    float w = __expf(v - m);
    atomicAdd(&den[d * H + hh], w);
    const float* hp = hsrc + (size_t)s * H * O + hh * O;
    float*       ap = agg  + (size_t)d * H * O + hh * O;
    #pragma unroll 4
    for (int o = 0; o < O; ++o) atomicAdd(&ap[o], w * hp[o]);
}

// ---------------- layer1 finalize: normalize + bias + ELU -> bf16 for next GEMM ----------------
__global__ void fin1_k(const float* __restrict__ agg, const float* __restrict__ den,
                       const float* __restrict__ b, __bf16* __restrict__ hb){
    int t = blockIdx.x * blockDim.x + threadIdx.x;
    if (t >= NN * H1DIM) return;
    int node = t >> 7, f = t & 127, hh = f >> 4;
    float v = agg[t] / (den[node * HEADS + hh] + 1e-16f) + b[f];
    v = v > 0.f ? v : (__expf(v) - 1.f);   // ELU
    hb[t] = (__bf16)v;
}

// ---------------- layer2 finalize: normalize + bias (heads=1, mean == identity) ----------------
__global__ void fin2_k(const float* __restrict__ agg, const float* __restrict__ den,
                       const float* __restrict__ b, float* __restrict__ y){
    int t = blockIdx.x * blockDim.x + threadIdx.x;
    if (t >= NN * CC) return;
    int node = t >> 4, o = t & 15;
    y[t] = agg[t] / (den[node] + 1e-16f) + b[o];
}

// ---------------- outputs: log_softmax(y), 1-cos(y,z), log_softmax(z) ----------------
__global__ void finish_k(const float* __restrict__ y, const float* __restrict__ z,
                         float* __restrict__ out){
    int n = blockIdx.x * blockDim.x + threadIdx.x;
    if (n >= NN) return;
    float yv[CC], zv[CC];
    #pragma unroll
    for (int o = 0; o < CC; ++o){ yv[o] = y[n * CC + o]; zv[o] = z[n * CC + o]; }
    float my = yv[0], mz = zv[0];
    #pragma unroll
    for (int o = 1; o < CC; ++o){ my = fmaxf(my, yv[o]); mz = fmaxf(mz, zv[o]); }
    float sy = 0.f, sz = 0.f;
    #pragma unroll
    for (int o = 0; o < CC; ++o){ sy += __expf(yv[o] - my); sz += __expf(zv[o] - mz); }
    float ly = __logf(sy), lz = __logf(sz);
    float dot = 0.f, n2y = 0.f, n2z = 0.f;
    #pragma unroll
    for (int o = 0; o < CC; ++o){ dot += yv[o]*zv[o]; n2y += yv[o]*yv[o]; n2z += zv[o]*zv[o]; }
    float ny = fmaxf(sqrtf(n2y), 1e-8f);
    float nz = fmaxf(sqrtf(n2z), 1e-8f);
    float cosv = dot / (ny * nz);
    float* o0 = out;                    // ls_y
    float* oc = out + (size_t)NN * CC;  // 1 - cos
    float* o2 = oc + NN;                // ls_z
    float* o3 = o2 + (size_t)NN * CC;   // ls_y
    float* o4 = o3 + (size_t)NN * CC;   // ls_y
    #pragma unroll
    for (int o = 0; o < CC; ++o){
        float a = yv[o] - my - ly;
        float bz = zv[o] - mz - lz;
        o0[n * CC + o] = a;
        o2[n * CC + o] = bz;
        o3[n * CC + o] = a;
        o4[n * CC + o] = a;
    }
    oc[n] = 1.f - cosv;
}

// ================= host side =================
static inline int cdiv(int a, int b){ return (a + b - 1) / b; }

struct Scratch {
    __bf16 *Wt1, *Wt2, *xb_hb;                 // xb aliases hb (bf16 [N,128])
    float  *ybuf, *zbuf;
    float  *h1, *agg1, *as1, *ad1, *den1;
    int    *mk1;
    // layer-2 overlays (live after layer-1 h1 region is dead)
    float  *h2, *agg2, *den2, *as2, *ad2;
    int    *mk2;
};

static void run_pass(const float* x, const int* ei,
                     const float* W1, const float* a1s, const float* a1d, const float* b1,
                     const float* W2, const float* a2s, const float* a2d, const float* b2,
                     float* yout, const Scratch& S, hipStream_t stream){
    const int T = 256;
    // cast input to bf16
    cast_bf16<<<cdiv(NN * FIN, T), T, 0, stream>>>(x, S.xb_hb, NN * FIN);
    // GEMM1: h1 = x @ W1   (WMMA bf16)
    gemm_n128<<<NN / 16, 256, 0, stream>>>(S.xb_hb, S.Wt1, S.h1);
    // attention coefficients, layer 1
    alphas_k<<<cdiv(NN * HEADS, T), T, 0, stream>>>(S.h1, a1s, a1d, S.as1, S.ad1, HEADS, HID);
    // init segment buffers
    fill_i32<<<cdiv(NN * HEADS, T), T, 0, stream>>>(S.mk1, (int)0x80000000, NN * HEADS);
    fill_f32<<<cdiv(NN * HEADS, T), T, 0, stream>>>(S.den1, 0.f, NN * HEADS);
    fill_f32<<<cdiv(NN * H1DIM, T), T, 0, stream>>>(S.agg1, 0.f, NN * H1DIM);
    // edge passes, layer 1
    int e1 = (EE + NN) * HEADS;
    edge_max_k<<<cdiv(e1, T), T, 0, stream>>>(ei, S.as1, S.ad1, S.mk1, HEADS);
    edge_acc_k<<<cdiv(e1, T), T, 0, stream>>>(ei, S.as1, S.ad1, S.mk1, S.h1, S.den1, S.agg1, HEADS, HID);
    // finalize layer 1 -> bf16 elu features (overwrites xb)
    fin1_k<<<cdiv(NN * H1DIM, T), T, 0, stream>>>(S.agg1, S.den1, b1, S.xb_hb);
    // GEMM2: h2 = elu(h1') @ W2   (WMMA bf16)
    gemm_n16<<<cdiv(NN / 16, 8), 256, 0, stream>>>(S.xb_hb, S.Wt2, S.h2, NN / 16);
    // attention coefficients, layer 2 (heads = 1)
    alphas_k<<<cdiv(NN, T), T, 0, stream>>>(S.h2, a2s, a2d, S.as2, S.ad2, 1, CC);
    fill_i32<<<cdiv(NN, T), T, 0, stream>>>(S.mk2, (int)0x80000000, NN);
    fill_f32<<<cdiv(NN, T), T, 0, stream>>>(S.den2, 0.f, NN);
    fill_f32<<<cdiv(NN * CC, T), T, 0, stream>>>(S.agg2, 0.f, NN * CC);
    int e2 = EE + NN;
    edge_max_k<<<cdiv(e2, T), T, 0, stream>>>(ei, S.as2, S.ad2, S.mk2, 1);
    edge_acc_k<<<cdiv(e2, T), T, 0, stream>>>(ei, S.as2, S.ad2, S.mk2, S.h2, S.den2, S.agg2, 1, CC);
    fin2_k<<<cdiv(NN * CC, T), T, 0, stream>>>(S.agg2, S.den2, b2, yout);
}

extern "C" void kernel_launch(void* const* d_in, const int* in_sizes, int n_in,
                              void* d_out, int out_size, void* d_ws, size_t ws_size,
                              hipStream_t stream){
    (void)in_sizes; (void)n_in; (void)out_size; (void)ws_size;
    const float* x1  = (const float*)d_in[0];
    const int*   ei1 = (const int*)  d_in[1];
    const float* x2  = (const float*)d_in[2];
    const int*   ei2 = (const int*)  d_in[3];
    const float* W1  = (const float*)d_in[4];
    const float* a1s = (const float*)d_in[5];
    const float* a1d = (const float*)d_in[6];
    const float* b1  = (const float*)d_in[7];
    const float* W2  = (const float*)d_in[8];
    const float* a2s = (const float*)d_in[9];
    const float* a2d = (const float*)d_in[10];
    const float* b2  = (const float*)d_in[11];

    char* ws = (char*)d_ws;
    size_t off = 0;
    auto bump = [&](size_t bytes) -> char* {
        char* p = ws + off;
        off += (bytes + 255) & ~(size_t)255;
        return p;
    };

    Scratch S;
    S.Wt1   = (__bf16*)bump((size_t)FIN * H1DIM * sizeof(__bf16));   // 32 KB
    S.Wt2   = (__bf16*)bump((size_t)H1DIM * CC * sizeof(__bf16));    //  4 KB
    S.ybuf  = (float*) bump((size_t)NN * CC * sizeof(float));        // 3.2 MB
    S.zbuf  = (float*) bump((size_t)NN * CC * sizeof(float));        // 3.2 MB
    S.xb_hb = (__bf16*)bump((size_t)NN * FIN * sizeof(__bf16));      // 12.8 MB (xb then elu-bf16)
    char* regC = bump((size_t)NN * H1DIM * sizeof(float));           // 25.6 MB (h1, then layer-2)
    S.agg1  = (float*) bump((size_t)NN * H1DIM * sizeof(float));     // 25.6 MB
    S.as1   = (float*) bump((size_t)NN * HEADS * sizeof(float));
    S.ad1   = (float*) bump((size_t)NN * HEADS * sizeof(float));
    S.den1  = (float*) bump((size_t)NN * HEADS * sizeof(float));
    S.mk1   = (int*)   bump((size_t)NN * HEADS * sizeof(int));

    S.h1   = (float*)regC;
    // layer-2 overlays inside regC (valid: h1 is dead before gemm_n16 writes h2)
    S.h2   = (float*)regC;
    S.agg2 = S.h2  + (size_t)NN * CC;
    S.den2 = S.agg2 + (size_t)NN * CC;
    S.as2  = S.den2 + NN;
    S.ad2  = S.as2  + NN;
    S.mk2  = (int*)(S.ad2 + NN);

    const int T = 256;
    // weight prep (pre-transposed bf16 so WMMA B k-pairs are contiguous)
    transpose_cast_bf16<<<cdiv(FIN * H1DIM, T), T, 0, stream>>>(W1, S.Wt1, FIN, H1DIM);
    transpose_cast_bf16<<<cdiv(H1DIM * CC, T), T, 0, stream>>>(W2, S.Wt2, H1DIM, CC);

    run_pass(x1, ei1, W1, a1s, a1d, b1, W2, a2s, a2d, b2, S.ybuf, S, stream);
    run_pass(x2, ei2, W1, a1s, a1d, b1, W2, a2s, a2d, b2, S.zbuf, S, stream);

    finish_k<<<cdiv(NN, T), T, 0, stream>>>(S.ybuf, S.zbuf, (float*)d_out);
}
